// MultiHeadAttentionCombinedQKV_15015205667433
// MI455X (gfx1250) — matrix operand order repaired and checked by reading, not compiled, not run on hardware
//
#include <hip/hip_runtime.h>
#include <hip/hip_fp16.h>

// Problem constants (from reference)
#define D_INN    1024
#define D_OUTT   1024
#define N3       3072          // 3*D_OUT
#define NHEADS   16
#define HEAD_DIM 64
#define BATCH    2
#define SEQ      2048
#define M_TOT    (BATCH*SEQ)   // 4096
#define SCALE_F  0.125f        // HEAD_DIM^-0.5

typedef __attribute__((ext_vector_type(16))) _Float16 v16h;
typedef __attribute__((ext_vector_type(8)))  float    v8f;

union Frag {               // one WMMA 16-bit operand: 8 VGPRs = 2x 128-bit loads
    v16h h;
    uint4 u[2];
};

static __device__ __forceinline__ v8f wmma_f16(v16h a, v16h b, v8f c) {
    // (neg_a, A, neg_b, B, c_mod, C, reuse_a, reuse_b)
    return __builtin_amdgcn_wmma_f32_16x16x32_f16(false, a, false, b, (short)0, c, false, false);
}

// ---------------------------------------------------------------- fp32 -> f16
__global__ void cvt_f32_f16(const float* __restrict__ in, _Float16* __restrict__ out, int n) {
    int i = blockIdx.x * blockDim.x + threadIdx.x;
    if (i < n) out[i] = (_Float16)in[i];
}

// ------------------------------------------------- QKV GEMM: C = X * Wqkv^T
// X: (4096,1024) f16 row-major.  W: (3072,1024) f16 row-major (out,in).
// One wave -> 32x64 output tile (8 accumulators, B-frags reused 2x).
// Block = 128 threads = 4 waves stacked in M -> 128x64 block tile.
__global__ __launch_bounds__(128) void gemm_qkv(const _Float16* __restrict__ A,
                                                const _Float16* __restrict__ W,
                                                _Float16* __restrict__ C) {
    const int lane = threadIdx.x & 31;
    const int wave = threadIdx.x >> 5;
    const int nl = lane & 15, h = lane >> 4;
    const int m0 = (blockIdx.y * 4 + wave) * 32;
    const int n0 = blockIdx.x * 64;

    v8f acc[2][4] = {};
    const _Float16* arow0 = A + (size_t)(m0 + nl) * D_INN;
    const _Float16* arow1 = A + (size_t)(m0 + 16 + nl) * D_INN;
    for (int k0 = 0; k0 < D_INN; k0 += 32) {
        Frag a0, a1;  // 16-bit A layout: runs [h*8,+8) and [16+h*8,+8)
        a0.u[0] = *(const uint4*)(arow0 + k0 + h * 8);
        a0.u[1] = *(const uint4*)(arow0 + k0 + 16 + h * 8);
        a1.u[0] = *(const uint4*)(arow1 + k0 + h * 8);
        a1.u[1] = *(const uint4*)(arow1 + k0 + 16 + h * 8);
#pragma unroll
        for (int t = 0; t < 4; ++t) {
            const _Float16* brow = W + (size_t)(n0 + t * 16 + nl) * D_INN;
            Frag b;  // B layout: runs [h*8,+8) and [16+h*8,+8)
            b.u[0] = *(const uint4*)(brow + k0 + h * 8);
            b.u[1] = *(const uint4*)(brow + k0 + 16 + h * 8);
            acc[0][t] = wmma_f16(a0.h, b.h, acc[0][t]);
            acc[1][t] = wmma_f16(a1.h, b.h, acc[1][t]);
        }
    }
#pragma unroll
    for (int mt = 0; mt < 2; ++mt)
#pragma unroll
        for (int t = 0; t < 4; ++t)
#pragma unroll
            for (int r = 0; r < 8; ++r) {
                const int row = m0 + mt * 16 + r + h * 8;   // C layout: M = r + h*8
                const int col = n0 + t * 16 + nl;           //           N = lane%16
                C[(size_t)row * N3 + col] = (_Float16)acc[mt][t][r];
            }
}

// ------------------------------------------------- Proj GEMM: out = CTX*Wp^T + b
__global__ __launch_bounds__(128) void gemm_proj(const _Float16* __restrict__ A,
                                                 const _Float16* __restrict__ W,
                                                 const float* __restrict__ bias,
                                                 float* __restrict__ C) {
    const int lane = threadIdx.x & 31;
    const int wave = threadIdx.x >> 5;
    const int nl = lane & 15, h = lane >> 4;
    const int m0 = (blockIdx.y * 4 + wave) * 32;
    const int n0 = blockIdx.x * 64;

    v8f acc[2][4] = {};
    const _Float16* arow0 = A + (size_t)(m0 + nl) * D_OUTT;
    const _Float16* arow1 = A + (size_t)(m0 + 16 + nl) * D_OUTT;
    for (int k0 = 0; k0 < D_OUTT; k0 += 32) {
        Frag a0, a1;
        a0.u[0] = *(const uint4*)(arow0 + k0 + h * 8);
        a0.u[1] = *(const uint4*)(arow0 + k0 + 16 + h * 8);
        a1.u[0] = *(const uint4*)(arow1 + k0 + h * 8);
        a1.u[1] = *(const uint4*)(arow1 + k0 + 16 + h * 8);
#pragma unroll
        for (int t = 0; t < 4; ++t) {
            const _Float16* brow = W + (size_t)(n0 + t * 16 + nl) * D_OUTT;
            Frag b;
            b.u[0] = *(const uint4*)(brow + k0 + h * 8);
            b.u[1] = *(const uint4*)(brow + k0 + 16 + h * 8);
            acc[0][t] = wmma_f16(a0.h, b.h, acc[0][t]);
            acc[1][t] = wmma_f16(a1.h, b.h, acc[1][t]);
        }
    }
#pragma unroll
    for (int t = 0; t < 4; ++t) {
        const int col = n0 + t * 16 + nl;
        const float bv = bias[col];
#pragma unroll
        for (int mt = 0; mt < 2; ++mt)
#pragma unroll
            for (int r = 0; r < 8; ++r) {
                const int row = m0 + mt * 16 + r + h * 8;
                C[(size_t)row * D_OUTT + col] = acc[mt][t][r] + bv;
            }
    }
}

// ------------------------------------------------- causal flash attention
// qkv: (B*S, 3, H, 64) f16.  ctx out: (B*S, H*64) f16.
// One 32-thread block (1 wave) per (b, head, 32-row q tile pair): K/V fragments
// and the staged V^T tile are shared by both 16-row q tiles.
__global__ __launch_bounds__(32) void attn(const _Float16* __restrict__ qkv,
                                           _Float16* __restrict__ ctx) {
    __shared__ __align__(16) _Float16 Pl[2][16][32];  // P tiles (qt, q x kv)
    __shared__ __align__(16) _Float16 VT[64][32];     // V^T tile (d x kv)

    const int lane = threadIdx.x;
    const int nl = lane & 15, h = lane >> 4;
    const int b = blockIdx.x >> 4, head = blockIdx.x & 15;
    const int q0 = blockIdx.y * 32;

    const _Float16* base = qkv + (size_t)b * SEQ * N3 + head * HEAD_DIM;

    // Q A-frags: two stacked 16-row tiles, K-dim 64 = two k-chunks of 32
    Frag qf[2][2];
#pragma unroll
    for (int qt = 0; qt < 2; ++qt) {
        const _Float16* qrow = base + (size_t)(q0 + qt * 16 + nl) * N3;  // Q slot (+0)
#pragma unroll
        for (int kc = 0; kc < 2; ++kc) {
            qf[qt][kc].u[0] = *(const uint4*)(qrow + kc * 32 + h * 8);
            qf[qt][kc].u[1] = *(const uint4*)(qrow + kc * 32 + 16 + h * 8);
        }
    }

    float mrow[2][8], lrow[2][8];
    v8f o[2][4] = {};
#pragma unroll
    for (int qt = 0; qt < 2; ++qt)
#pragma unroll
        for (int r = 0; r < 8; ++r) { mrow[qt][r] = -1e30f; lrow[qt][r] = 0.0f; }

    const int jmax = (q0 + 31) >> 5;           // last 32-wide kv block touching diag
    for (int j = 0; j <= jmax; ++j) {
        const int kv0 = j * 32;

        // ---- scores S = Q K^T; each K B-frag feeds both q tiles
        v8f s[2][2] = {};
#pragma unroll
        for (int t = 0; t < 2; ++t) {
            const _Float16* krow = base + D_OUTT + (size_t)(kv0 + t * 16 + nl) * N3;  // K slot
#pragma unroll
            for (int kc = 0; kc < 2; ++kc) {
                Frag bk;
                bk.u[0] = *(const uint4*)(krow + kc * 32 + h * 8);
                bk.u[1] = *(const uint4*)(krow + kc * 32 + 16 + h * 8);
                s[0][t] = wmma_f16(qf[0][kc].h, bk.h, s[0][t]);
                s[1][t] = wmma_f16(qf[1][kc].h, bk.h, s[1][t]);
            }
        }

        // ---- stage V^T in LDS once per kv block: lane owns kv row (kv0+lane)
        {
            const _Float16* vrow = base + 2 * D_OUTT + (size_t)(kv0 + lane) * N3;  // V slot
#pragma unroll
            for (int c8 = 0; c8 < 8; ++c8) {
                union { uint4 u; _Float16 f[8]; } v;
                v.u = *(const uint4*)(vrow + c8 * 8);
#pragma unroll
                for (int e = 0; e < 8; ++e) VT[c8 * 8 + e][lane] = v.f[e];
            }
        }

        // ---- mask + scale + online softmax per q tile (row = r + h*8, col = lane%16)
        float alpha[2][8];
#pragma unroll
        for (int qt = 0; qt < 2; ++qt)
#pragma unroll
            for (int r = 0; r < 8; ++r) {
                const int rowg = q0 + qt * 16 + r + h * 8;
                float s0 = (kv0 + nl > rowg)      ? -1e30f : s[qt][0][r] * SCALE_F;
                float s1 = (kv0 + 16 + nl > rowg) ? -1e30f : s[qt][1][r] * SCALE_F;
                float mx = fmaxf(s0, s1);
                mx = fmaxf(mx, __shfl_xor(mx, 1, 32));
                mx = fmaxf(mx, __shfl_xor(mx, 2, 32));
                mx = fmaxf(mx, __shfl_xor(mx, 4, 32));
                mx = fmaxf(mx, __shfl_xor(mx, 8, 32));
                const float mnew = fmaxf(mrow[qt][r], mx);
                const float a_ = __expf(mrow[qt][r] - mnew);
                const float e0 = __expf(s0 - mnew);
                const float e1 = __expf(s1 - mnew);
                float sum = e0 + e1;
                sum += __shfl_xor(sum, 1, 32);
                sum += __shfl_xor(sum, 2, 32);
                sum += __shfl_xor(sum, 4, 32);
                sum += __shfl_xor(sum, 8, 32);
                lrow[qt][r] = lrow[qt][r] * a_ + sum;
                mrow[qt][r] = mnew;
                alpha[qt][r] = a_;
                Pl[qt][r + h * 8][nl]      = (_Float16)e0;
                Pl[qt][r + h * 8][16 + nl] = (_Float16)e1;
            }
        __syncthreads();

        // ---- P as A-frags; each V^T B-frag feeds both q tiles; O = O*alpha + P*V
        Frag pf[2];
#pragma unroll
        for (int qt = 0; qt < 2; ++qt) {
            pf[qt].u[0] = *(const uint4*)&Pl[qt][nl][h * 8];
            pf[qt].u[1] = *(const uint4*)&Pl[qt][nl][16 + h * 8];
        }
#pragma unroll
        for (int c = 0; c < 4; ++c) {
            Frag bv;
            bv.u[0] = *(const uint4*)&VT[c * 16 + nl][h * 8];
            bv.u[1] = *(const uint4*)&VT[c * 16 + nl][16 + h * 8];
#pragma unroll
            for (int qt = 0; qt < 2; ++qt) {
                v8f oc = o[qt][c];
#pragma unroll
                for (int r = 0; r < 8; ++r) oc[r] *= alpha[qt][r];
                o[qt][c] = wmma_f16(pf[qt].h, bv.h, oc);
            }
        }
        __syncthreads();  // protect LDS reuse next iteration
    }

    // ---- normalize and write ctx (b, s, head*64 + d)
#pragma unroll
    for (int qt = 0; qt < 2; ++qt) {
        float inv[8];
#pragma unroll
        for (int r = 0; r < 8; ++r) inv[r] = 1.0f / lrow[qt][r];
#pragma unroll
        for (int c = 0; c < 4; ++c)
#pragma unroll
            for (int r = 0; r < 8; ++r) {
                const int row = q0 + qt * 16 + r + h * 8;
                ctx[((size_t)b * SEQ + row) * D_OUTT + head * HEAD_DIM + c * 16 + nl] =
                    (_Float16)(o[qt][c][r] * inv[r]);
            }
    }
}

// ----------------------------------------------------------------- launcher
extern "C" void kernel_launch(void* const* d_in, const int* in_sizes, int n_in,
                              void* d_out, int out_size, void* d_ws, size_t ws_size,
                              hipStream_t stream) {
    const float* x      = (const float*)d_in[0];   // (2,2048,1024)
    const float* w_qkv  = (const float*)d_in[1];   // (3072,1024)
    const float* w_proj = (const float*)d_in[2];   // (1024,1024)
    const float* b_proj = (const float*)d_in[3];   // (1024,)

    char* ws = (char*)d_ws;
    _Float16* Xh   = (_Float16*)(ws);                         //  8 MB
    _Float16* Wqh  = (_Float16*)(ws + 8u  * 1024 * 1024);     //  6 MB
    _Float16* Wph  = (_Float16*)(ws + 14u * 1024 * 1024);     //  2 MB
    _Float16* QKVh = (_Float16*)(ws + 16u * 1024 * 1024);     // 24 MB
    _Float16* CTXh = (_Float16*)(ws + 40u * 1024 * 1024);     //  8 MB

    const int nX  = M_TOT * D_INN;     // 4194304
    const int nWq = N3 * D_INN;        // 3145728
    const int nWp = D_OUTT * D_INN;    // 1048576
    cvt_f32_f16<<<(nX  + 255) / 256, 256, 0, stream>>>(x,      Xh,  nX);
    cvt_f32_f16<<<(nWq + 255) / 256, 256, 0, stream>>>(w_qkv,  Wqh, nWq);
    cvt_f32_f16<<<(nWp + 255) / 256, 256, 0, stream>>>(w_proj, Wph, nWp);

    // QKV projection: (4096 x 3072) = X * Wqkv^T
    gemm_qkv<<<dim3(N3 / 64, M_TOT / 128), 128, 0, stream>>>(Xh, Wqh, QKVh);

    // causal attention: 32 (b,h) x 64 q-tile-pairs
    attn<<<dim3(BATCH * NHEADS, SEQ / 32), 32, 0, stream>>>(QKVh, CTXh);

    // output projection + bias -> fp32 d_out
    gemm_proj<<<dim3(D_OUTT / 64, M_TOT / 128), 128, 0, stream>>>(CTXh, Wph, b_proj, (float*)d_out);
}
